// DeeperDynamicEdgeNetPredictFlow_44358422233177
// MI455X (gfx1250) — matrix operand order repaired and checked by reading, not compiled, not run on hardware
//
#include <hip/hip_runtime.h>
#include <hip/hip_bf16.h>

// ---------------------------------------------------------------------------
// Types for CDNA5 WMMA (wave32, 16x16x32 f16 -> f32 accumulate)
// ---------------------------------------------------------------------------
typedef _Float16 half_t;
typedef _Float16 v16h __attribute__((ext_vector_type(16)));
typedef _Float16 v8h  __attribute__((ext_vector_type(8)));
typedef float    v8f  __attribute__((ext_vector_type(8)));

// ---------------------------------------------------------------------------
// GEMM + fused bias + fused BN column-statistics epilogue.
//   H[m][n] = sum_k A[m][k] * W[n][k] + bias[n]
//   sums[n]   += sum_m H[m][n]        (via LDS ds_add_f32 then global atomics)
//   sums[N+n] += sum_m H[m][n]^2
// A: [M, Kp] f16 row-major (Kp % 32 == 0, zero padded)
// W: [N, Kp] f16 row-major (packed weights, zero padded)
// Block = 8 waves. All waves share the same NT*16 output columns, so the
// weight panel (NT*16 x Kp, <= 36 KB) is staged once into LDS and read back
// with ds_load_b128. Each wave computes MT=2 (32) rows x NT*16 columns.
// Requires M % 256 == 0 (true for all call sites: 524288, 262144).
// ---------------------------------------------------------------------------
template <int NT>
__global__ __launch_bounds__(256) void gemm_wmma(
    const half_t* __restrict__ A, const half_t* __restrict__ W,
    const float* __restrict__ bias, float* __restrict__ H,
    float* __restrict__ sums, int M, int N, int Kp)
{
    constexpr int MT = 2;
    extern __shared__ char smem_raw[];
    half_t* sW = (half_t*)smem_raw;
    const int panelElems = NT * 16 * Kp;
    float* sstat = (float*)(smem_raw + (((size_t)panelElems * 2 + 15) & ~(size_t)15));

    const int lane = threadIdx.x & 31;
    const int wave = threadIdx.x >> 5;
    const int m0   = blockIdx.x * 256 + wave * (MT * 16);
    const int n0   = blockIdx.y * (NT * 16);

    // ---- stage weight panel (rows n0 .. n0+NT*16-1) into LDS ----
    {
        const half_t* gW = W + (size_t)n0 * (size_t)Kp;
        for (int i = threadIdx.x * 8; i < panelElems; i += 256 * 8)
            *(v8h*)(sW + i) = *(const v8h*)(gW + i);
    }
    // zero the LDS stats accumulators
    for (int i = threadIdx.x; i < 2 * NT * 16; i += 256) sstat[i] = 0.0f;
    __syncthreads();

    const int hsel = lane >> 4;    // 0: lanes 0-15, 1: lanes 16-31
    const int l16  = lane & 15;

    size_t rowA[MT];
#pragma unroll
    for (int mt = 0; mt < MT; ++mt)
        rowA[mt] = (size_t)(m0 + mt * 16 + l16) * (size_t)Kp;

    v8f acc[MT][NT] = {};

    for (int k0 = 0; k0 < Kp; k0 += 32) {
        // A 16x32 tiles: lanes 0-15 rows +0..15, K {k0..k0+15}
        //                lanes 16-31 same rows, K {k0+16..k0+31}
        const int ka = k0 + hsel * 16;
        v16h av[MT];
#pragma unroll
        for (int mt = 0; mt < MT; ++mt) {
            v8h alo = *(const v8h*)(A + rowA[mt] + ka);
            v8h ahi = *(const v8h*)(A + rowA[mt] + ka + 8);
#pragma unroll
            for (int i = 0; i < 8; ++i) { av[mt][i] = alo[i]; av[mt][i + 8] = ahi[i]; }
        }
        // B 32x16 tiles from LDS: lane holds column n_local = t*16+l16,
        // K = k0 + hsel*16 .. +15 (16 contiguous f16)
        const int kb = k0 + hsel * 16;
#pragma unroll
        for (int t = 0; t < NT; ++t) {
            v16h bv = *(const v16h*)(sW + (size_t)(t * 16 + l16) * Kp + kb);
#pragma unroll
            for (int mt = 0; mt < MT; ++mt)
                acc[mt][t] = __builtin_amdgcn_wmma_f32_16x16x32_f16(
                    false, av[mt], false, bv, (short)0, acc[mt][t], false, false);
        }
    }

    // ---- epilogue: bias, store H, accumulate column sums/sumsq ----
    // C/D layout: VGPR r -> M = tile_m + hsel*8 + r, N = n0 + t*16 + l16
#pragma unroll
    for (int t = 0; t < NT; ++t) {
        const int n  = n0 + t * 16 + l16;
        const float bv = bias[n];
#pragma unroll
        for (int mt = 0; mt < MT; ++mt) {
            float s = 0.0f, s2 = 0.0f;
#pragma unroll
            for (int r = 0; r < 8; ++r) {
                const int m = m0 + mt * 16 + hsel * 8 + r;
                float hv = acc[mt][t][r] + bv;
                H[(size_t)m * N + n] = hv;
                s += hv; s2 += hv * hv;
            }
            atomicAdd(&sstat[t * 16 + l16], s);
            atomicAdd(&sstat[NT * 16 + t * 16 + l16], s2);
        }
    }
    __syncthreads();
    if (threadIdx.x < (unsigned)(NT * 16)) {
        int i = threadIdx.x;
        atomicAdd(&sums[n0 + i],     sstat[i]);
        atomicAdd(&sums[N + n0 + i], sstat[NT * 16 + i]);
    }
}

// ---------------------------------------------------------------------------
// kNN: one workgroup per graph, features staged in LDS (<=139KB of 320KB WGP LDS)
// Each thread owns one node row, keeps a sorted top-16 by squared distance.
// ---------------------------------------------------------------------------
__global__ void knn_kernel(const half_t* __restrict__ feat, int stride, int D,
                           int N, int* __restrict__ idx)
{
    extern __shared__ float sf[];
    const int b = blockIdx.x;
    const int tid = threadIdx.x;
    const int total = N * D;
    for (int i = tid; i < total; i += blockDim.x) {
        int node = i / D, c = i - node * D;
        sf[i] = (float)feat[(size_t)(b * N + node) * stride + c];
    }
    __syncthreads();
    if (tid >= N) return;

    const float* xi = sf + tid * D;
    float bd[16]; int bi[16];
#pragma unroll
    for (int s = 0; s < 16; ++s) { bd[s] = 3.4e38f; bi[s] = 0; }

    for (int j = 0; j < N; ++j) {
        const float* xj = sf + j * D;
        float d = 0.0f;
        for (int t = 0; t < D; ++t) { float df = xi[t] - xj[t]; d += df * df; }
        if (d < bd[15]) {
            bd[15] = d; bi[15] = j;
#pragma unroll
            for (int s = 15; s > 0; --s) {
                if (bd[s] < bd[s - 1]) {
                    float td = bd[s]; bd[s] = bd[s - 1]; bd[s - 1] = td;
                    int   ti = bi[s]; bi[s] = bi[s - 1]; bi[s - 1] = ti;
                } else break;
            }
        }
    }
#pragma unroll
    for (int s = 0; s < 16; ++s) idx[(size_t)(b * N + tid) * 16 + s] = bi[s];
}

// ---------------------------------------------------------------------------
// Build edge features e[r] = [x_i, x_j - x_i] padded to Kp with zeros, f16.
// r = node*16 + k. rows = B*N*16.
// ---------------------------------------------------------------------------
__global__ void gather_edge_feat(const half_t* __restrict__ feat, int stride, int D,
                                 const int* __restrict__ idx, int N,
                                 half_t* __restrict__ e, int Kp, int rows)
{
    int gid = blockIdx.x * blockDim.x + threadIdx.x;
    int tot = rows * Kp;
    if (gid >= tot) return;
    int r = gid / Kp;
    int c = gid - r * Kp;
    int node = r >> 4;
    int jl = idx[r];
    int jg = (node / N) * N + jl;
    float v = 0.0f;
    if (c < D)            v = (float)feat[(size_t)node * stride + c];
    else if (c < 2 * D) { int cc = c - D;
                          v = (float)feat[(size_t)jg * stride + cc]
                            - (float)feat[(size_t)node * stride + cc]; }
    e[gid] = (half_t)v;
}

// ---------------------------------------------------------------------------
// Final edge features e[r] = [xc[row[r]], xc[col[r]]] padded to Kp, f16.
// edge_index is int64 [2, E].
// ---------------------------------------------------------------------------
__global__ void gather_out_feat(const half_t* __restrict__ xc, int stride, int D,
                                const long long* __restrict__ ei, int E,
                                half_t* __restrict__ e, int Kp)
{
    int gid = blockIdx.x * blockDim.x + threadIdx.x;
    int tot = E * Kp;
    if (gid >= tot) return;
    int r = gid / Kp;
    int c = gid - r * Kp;
    float v = 0.0f;
    if (c < D)          { long long n = ei[r];             v = (float)xc[(size_t)n * stride + c]; }
    else if (c < 2 * D) { long long n = ei[(size_t)E + r]; v = (float)xc[(size_t)n * stride + (c - D)]; }
    e[gid] = (half_t)v;
}

// ---------------------------------------------------------------------------
// BatchNorm (training mode) support kernels
// ---------------------------------------------------------------------------
__global__ void zerof(float* __restrict__ p, int n)
{
    int i = blockIdx.x * blockDim.x + threadIdx.x;
    if (i < n) p[i] = 0.0f;
}

// used only for bn0 over the raw f32 input (blockDim.x == N)
__global__ void col_stats(const float* __restrict__ H, int M, int N, int rowsPer,
                          float* __restrict__ sums)
{
    int c = threadIdx.x;
    int r0 = blockIdx.x * rowsPer;
    int r1 = min(M, r0 + rowsPer);
    float s = 0.0f, s2 = 0.0f;
    for (int r = r0; r < r1; ++r) {
        float v = H[(size_t)r * N + c];
        s += v; s2 += v * v;
    }
    atomicAdd(&sums[c], s);
    atomicAdd(&sums[N + c], s2);
}

__global__ void bn_scale(const float* __restrict__ sums, int M, int N,
                         const float* __restrict__ g, const float* __restrict__ be,
                         float* __restrict__ sc)
{
    int c = blockIdx.x * blockDim.x + threadIdx.x;
    if (c >= N) return;
    float inv_m = 1.0f / (float)M;
    float mean = sums[c] * inv_m;
    float var  = sums[N + c] * inv_m - mean * mean;
    float s    = g[c] * rsqrtf(var + 1e-5f);
    sc[c]     = s;
    sc[N + c] = be[c] - mean * s;
}

// out[r*ostride + ooff + c] = f16( relu?( H[r][c]*scale[c]+shift[c] ) )
__global__ void bn_apply(const float* __restrict__ H, int M, int N,
                         const float* __restrict__ sc, half_t* __restrict__ out,
                         int ostride, int ooff, int relu)
{
    int gid = blockIdx.x * blockDim.x + threadIdx.x;
    int tot = M * N;
    if (gid >= tot) return;
    int r = gid / N;
    int c = gid - r * N;
    float v = H[gid] * sc[c] + sc[N + c];
    if (relu) v = fmaxf(v, 0.0f);
    out[(size_t)r * ostride + ooff + c] = (half_t)v;
}

// ---------------------------------------------------------------------------
// Misc small kernels
// ---------------------------------------------------------------------------
__global__ void f32_to_f16(const float* __restrict__ in, half_t* __restrict__ out, int n)
{
    int i = blockIdx.x * blockDim.x + threadIdx.x;
    if (i < n) out[i] = (half_t)in[i];
}

// Pack W[N][K] f32 -> Wp[N][Kp] f16, zero-padded along K
__global__ void pack_w(const float* __restrict__ W, half_t* __restrict__ Wp,
                       int N, int K, int Kp)
{
    int gid = blockIdx.x * blockDim.x + threadIdx.x;
    int tot = N * Kp;
    if (gid >= tot) return;
    int n = gid / Kp;
    int k = gid - n * Kp;
    Wp[gid] = (half_t)((k < K) ? W[(size_t)n * K + k] : 0.0f);
}

// mean over K=16 neighbors -> write into xc at column offset 4
__global__ void mean_k(const half_t* __restrict__ act, int D,
                       half_t* __restrict__ xc, int stride, int nodes)
{
    int gid = blockIdx.x * blockDim.x + threadIdx.x;
    int tot = nodes * D;
    if (gid >= tot) return;
    int node = gid / D;
    int c = gid - node * D;
    float s = 0.0f;
#pragma unroll
    for (int k = 0; k < 16; ++k)
        s += (float)act[(size_t)(node * 16 + k) * D + c];
    xc[(size_t)node * stride + 4 + c] = (half_t)(s * 0.0625f);
}

// final 256 -> 1 projection (memory bound; scalar path is fine)
__global__ void out_dot(const half_t* __restrict__ act, const float* __restrict__ W3,
                        const float* __restrict__ b3, float* __restrict__ out, int E)
{
    int e = blockIdx.x * blockDim.x + threadIdx.x;
    if (e >= E) return;
    const half_t* a = act + (size_t)e * 256;
    float s = b3[0];
    for (int c = 0; c < 256; ++c) s += (float)a[c] * W3[c];
    out[e] = s;
}

// ---------------------------------------------------------------------------
static void launch_gemm(const half_t* A, const half_t* W, const float* bias,
                        float* H, float* sums, int M, int N, int Kp, hipStream_t s)
{
    dim3 blk(256);
    int mtiles = (M + 255) / 256;   // MT=2: 256 rows per block
    if (N % 64 == 0) {
        constexpr int NT = 4;
        size_t shmem = ((size_t)NT * 16 * Kp * 2 + 15 & ~(size_t)15) + 2 * NT * 16 * 4;
        dim3 grd(mtiles, N / 64);
        gemm_wmma<NT><<<grd, blk, shmem, s>>>(A, W, bias, H, sums, M, N, Kp);
    } else {
        constexpr int NT = 2;
        size_t shmem = ((size_t)NT * 16 * Kp * 2 + 15 & ~(size_t)15) + 2 * NT * 16 * 4;
        dim3 grd(mtiles, N / 32);
        gemm_wmma<NT><<<grd, blk, shmem, s>>>(A, W, bias, H, sums, M, N, Kp);
    }
}

extern "C" void kernel_launch(void* const* d_in, const int* in_sizes, int n_in,
                              void* d_out, int out_size, void* d_ws, size_t ws_size,
                              hipStream_t stream)
{
    (void)n_in; (void)out_size; (void)ws_size;

    const float*     x  = (const float*)d_in[0];
    const long long* ei = (const long long*)d_in[1];

    const int Bg  = 64;
    const int BN  = in_sizes[0] / 4;      // 32768
    const int Np  = BN / Bg;              // 512
    const int BNK = BN * 16;              // 524288
    const int E   = in_sizes[1] / 2;      // 262144

    // ---- workspace layout (bump allocator, 256B aligned) ----
    char* ws = (char*)d_ws;
    size_t off = 0;
    auto alloc = [&](size_t bytes) -> char* {
        off = (off + 255) & ~(size_t)255;
        char* p = ws + off;
        off += bytes;
        return p;
    };
    const int XCS = 144;  // xc row stride (holds up to 132 cols)
    half_t* XC   = (half_t*)alloc((size_t)BN * XCS * sizeof(half_t));
    half_t* F1   = (half_t*)alloc((size_t)BN * 4 * sizeof(half_t));
    int*    IDX  = (int*)  alloc((size_t)BN * 16 * sizeof(int));
    float*  SUMS = (float*)alloc(512 * sizeof(float));
    float*  SC   = (float*)alloc(512 * sizeof(float));

    // layers: conv1 (0..2), conv2 (3..5), conv3 (6..8), out (9..10)
    static const int LN [11] = {32,32,32,  64,64,64,  128,128,128, 256,256};
    static const int LK [11] = { 8,32,32,  72,64,64,  136,128,128, 264,256};
    static const int LKP[11] = {32,32,32,  96,64,64,  160,128,128, 288,256};
    static const int LIX[11] = { 5, 9,13,  17,21,25,   29, 33, 37,  41, 45};
    half_t* WP[11];
    for (int i = 0; i < 11; ++i)
        WP[i] = (half_t*)alloc((size_t)LN[i] * LKP[i] * sizeof(half_t));

    half_t* AR = (half_t*)alloc((size_t)BNK * 160 * sizeof(half_t)); // edge feats / act ping
    float*  HB = (float*) alloc((size_t)BNK * 128 * sizeof(float));  // pre-activation f32
    half_t* CB = (half_t*)alloc((size_t)BNK * 128 * sizeof(half_t)); // act pong

    // ---- pack all weights to padded f16 ----
    for (int i = 0; i < 11; ++i) {
        int tot = LN[i] * LKP[i];
        pack_w<<<(tot + 255) / 256, 256, 0, stream>>>(
            (const float*)d_in[LIX[i]], WP[i], LN[i], LK[i], LKP[i]);
    }

    // ---- x -> f16 features for conv1 kNN / gather ----
    f32_to_f16<<<(BN * 4 + 255) / 256, 256, 0, stream>>>(x, F1, BN * 4);

    // ---- bn0 -> x1 into XC cols 0..3 (no ReLU) ----
    zerof<<<1, 512, 0, stream>>>(SUMS, 8);
    col_stats<<<(BN + 511) / 512, 4, 0, stream>>>(x, BN, 4, 512, SUMS);
    bn_scale<<<1, 4, 0, stream>>>(SUMS, BN, 4,
                                  (const float*)d_in[3], (const float*)d_in[4], SC);
    bn_apply<<<(BN * 4 + 255) / 256, 256, 0, stream>>>(x, BN, 4, SC, XC, XCS, 0, 0);

    // ---- Linear + BN(train) + ReLU layer (stats fused in GEMM epilogue) ----
    auto mlp_layer = [&](const half_t* Ain, int L, int M, half_t* Aout,
                         int ostride, int relu) {
        int N = LN[L], Kp = LKP[L];
        zerof<<<1, 512, 0, stream>>>(SUMS, 2 * N);
        launch_gemm(Ain, WP[L], (const float*)d_in[LIX[L] + 1], HB, SUMS,
                    M, N, Kp, stream);
        bn_scale<<<1, N, 0, stream>>>(SUMS, M, N,
                                      (const float*)d_in[LIX[L] + 2],
                                      (const float*)d_in[LIX[L] + 3], SC);
        int tot = M * N;
        bn_apply<<<(tot + 255) / 256, 256, 0, stream>>>(HB, M, N, SC, Aout,
                                                        ostride, 0, relu);
    };

    // ---- one DynamicEdgeConv stage ----
    auto conv_stage = [&](const half_t* feat, int fstride, int D, int L0) {
        knn_kernel<<<Bg, Np, (size_t)Np * D * sizeof(float), stream>>>(
            feat, fstride, D, Np, IDX);
        int Kpe = LKP[L0];
        int tot = BNK * Kpe;
        gather_edge_feat<<<(tot + 255) / 256, 256, 0, stream>>>(
            feat, fstride, D, IDX, Np, AR, Kpe, BNK);
        int Dout = LN[L0];
        mlp_layer(AR, L0,     BNK, CB, Dout, 1);
        mlp_layer(CB, L0 + 1, BNK, AR, Dout, 1);
        mlp_layer(AR, L0 + 2, BNK, CB, Dout, 1);
        int totm = BN * Dout;
        mean_k<<<(totm + 255) / 256, 256, 0, stream>>>(CB, Dout, XC, XCS, BN);
    };

    conv_stage(F1, 4,   4,  0);   // conv1: kNN on raw x
    conv_stage(XC, XCS, 36, 3);   // conv2: kNN on [x1, h1]
    conv_stage(XC, XCS, 68, 6);   // conv3: kNN on [x1, h2]

    // ---- output MLP over edges ----
    {
        int tot = E * 288;
        gather_out_feat<<<(tot + 255) / 256, 256, 0, stream>>>(
            XC, XCS, 132, ei, E, AR, 288);
        mlp_layer(AR, 9,  E, CB, 256, 1);
        mlp_layer(CB, 10, E, AR, 256, 1);
        out_dot<<<(E + 255) / 256, 256, 0, stream>>>(
            AR, (const float*)d_in[49], (const float*)d_in[50], (float*)d_out, E);
    }
}
